// ISEFWINNER_SMALL_1090921693506
// MI455X (gfx1250) — compile-verified, matrix operand order checked by reading, not run on hardware
//
#include <hip/hip_runtime.h>
#include <hip/hip_bf16.h>
#include <math.h>

typedef __attribute__((ext_vector_type(16))) _Float16 v16h;
typedef __attribute__((ext_vector_type(8)))  _Float16 v8h;
typedef __attribute__((ext_vector_type(8)))  float    v8f;
typedef __attribute__((ext_vector_type(4)))  float    v4f;

#define BATCH 4096
#define SEQ   256
#define TLEN  256
#define HID   40
#define NLAY  4
#define NG    160   // 4*HID

// ---------------- LDS layout for the fused stacked-LSTM kernel ----------------
// halves: wih[4][160][64] | whh[4][160][64] | h[2][4][16][64] | x[16][32]
// floats: per-wave gate scratch [10][16][16]
#define H_WIH   0
#define N_WIH   (4*160*64)
#define H_WHH   (H_WIH + N_WIH)
#define N_WHH   (4*160*64)
#define H_H     (H_WHH + N_WHH)
#define N_H     (2*4*16*64)
#define H_X     (H_H + N_H)
#define N_X     (16*32)
#define HALFS_TOTAL (H_X + N_X)
#define N_SCR   (10*16*16)
#define LSTM_SMEM_BYTES (HALFS_TOTAL*2 + N_SCR*4)   // ~187 KB < 320 KB WGP LDS

// ---- branch-free activations on the CDNA5 transcendental unit ----
__device__ __forceinline__ float fast_tanh(float x) {
#if __has_builtin(__builtin_amdgcn_tanhf)
    return __builtin_amdgcn_tanhf(x);                 // native V_TANH_F32
#else
    // tanh(x) = 2/(1+exp(-2x)) - 1, exp via raw V_EXP_F32 (no range fixup)
    return 2.0f * __builtin_amdgcn_rcpf(
               1.0f + __builtin_amdgcn_exp2f(-2.8853900817779268f * x)) - 1.0f;
#endif
}
__device__ __forceinline__ float fast_sig(float x) {
#if __has_builtin(__builtin_amdgcn_tanhf)
    return 0.5f * __builtin_amdgcn_tanhf(0.5f * x) + 0.5f;
#else
    return __builtin_amdgcn_rcpf(1.0f + __builtin_amdgcn_exp2f(-1.4426950408889634f * x));
#endif
}

// Load one 16x32 f16 WMMA fragment (A or B) from an LDS row-major tile.
// Per ISA layout: lanes 0-15 hold K {0..7,16..23}, lanes 16-31 hold K {8..15,24..31},
// row/col = lane & 15.  Two 16-byte LDS reads per lane.
__device__ __forceinline__ v16h frag16(const _Float16* base, int stride, int lane, int koff) {
    const int r  = lane & 15;
    const int kb = ((lane >> 4) << 3) + koff;
    const _Float16* p = base + r * stride + kb;
    union { v16h v; v8h h[2]; } u;
    u.h[0] = *(const v8h*)(p);
    u.h[1] = *(const v8h*)(p + 16);
    return u.v;
}

__device__ __forceinline__ v8f wmma_acc(v16h a, v16h b, v8f c) {
    return __builtin_amdgcn_wmma_f32_16x16x32_f16(false, a, false, b, (short)0, c, false, false);
}

// gate-interleaved weight row permutation: original row n = g*40 + j
// -> physical row 16*(j/4) + 4*(j%4) + g, so wave w's 16 WMMA columns hold
// complete (i,f,g,o) quads for hidden units j = 4w..4w+3.
__device__ __forceinline__ int permute_row(int n_orig) {
    int g = n_orig / HID, j = n_orig % HID;
    return 16 * (j >> 2) + 4 * (j & 3) + g;
}

// =============================================================================
// Kernel 1: MLP1 on X1/X2.  One thread per (b,s) row.  Writes f16 features
// pre-transposed to [t][b][16] (15 valid + 1 zero pad) so the LSTM kernel can
// stage a contiguous 16x16-half A-tile per timestep.
// =============================================================================
__global__ __launch_bounds__(256) void mlp1_kernel(
    const float* __restrict__ X1, const float* __restrict__ X2,
    const float* __restrict__ w1, const float* __restrict__ b1,
    const float* __restrict__ w2, const float* __restrict__ b2,
    const float* __restrict__ w3, const float* __restrict__ b3,
    const float* __restrict__ w4, const float* __restrict__ b4,
    _Float16* __restrict__ f1, _Float16* __restrict__ f2)
{
    __shared__ float s[4595];
    const int tid = threadIdx.x;
    for (int i = tid; i < 250;  i += 256) s[i]        = w1[i];
    for (int i = tid; i < 50;   i += 256) s[250 + i]  = b1[i];
    for (int i = tid; i < 2000; i += 256) s[300 + i]  = w2[i];
    for (int i = tid; i < 40;   i += 256) s[2300 + i] = b2[i];
    for (int i = tid; i < 1600; i += 256) s[2340 + i] = w3[i];
    for (int i = tid; i < 40;   i += 256) s[3940 + i] = b3[i];
    for (int i = tid; i < 600;  i += 256) s[3980 + i] = w4[i];
    for (int i = tid; i < 15;   i += 256) s[4580 + i] = b4[i];
    __syncthreads();

    const float* sw1 = s;        const float* sb1 = s + 250;
    const float* sw2 = s + 300;  const float* sb2 = s + 2300;
    const float* sw3 = s + 2340; const float* sb3 = s + 3940;
    const float* sw4 = s + 3980; const float* sb4 = s + 4580;

    const int r = blockIdx.x * 256 + tid;           // r in [0, BATCH*SEQ)
    const float* X = blockIdx.y ? X2 : X1;
    _Float16* dst  = blockIdx.y ? f2 : f1;

    float x[5];
    #pragma unroll
    for (int k = 0; k < 5; ++k) x[k] = X[(size_t)r * 5 + k];

    float h1[50];
    #pragma unroll
    for (int j = 0; j < 50; ++j) {
        float a = sb1[j];
        #pragma unroll
        for (int k = 0; k < 5; ++k) a = fmaf(sw1[j * 5 + k], x[k], a);
        h1[j] = fmaxf(a, 0.0f);
    }
    float h2[40];
    #pragma unroll 4
    for (int j = 0; j < 40; ++j) {
        float a = sb2[j];
        #pragma unroll
        for (int k = 0; k < 50; ++k) a = fmaf(sw2[j * 50 + k], h1[k], a);
        h2[j] = fast_tanh(a);
    }
    float h3[40];
    #pragma unroll 4
    for (int j = 0; j < 40; ++j) {
        float a = sb3[j];
        #pragma unroll
        for (int k = 0; k < 40; ++k) a = fmaf(sw3[j * 40 + k], h2[k], a);
        h3[j] = fast_tanh(a);
    }
    const int b    = r >> 8;           // SEQ == 256
    const int srow = r & 255;
    _Float16* o = dst + ((size_t)srow * BATCH + b) * 16;
    #pragma unroll
    for (int j = 0; j < 15; ++j) {
        float a = sb4[j];
        #pragma unroll
        for (int k = 0; k < 40; ++k) a = fmaf(sw4[j * 40 + k], h3[k], a);
        o[j] = (_Float16)a;
    }
    o[15] = (_Float16)0.0f;
}

// =============================================================================
// Kernel 2: fused 4-layer stacked LSTM, WMMA-based.
// One workgroup = 16 batch rows (M=16) x 10 waves (gate-interleaved N-tiles).
// t outer, layer inner: weight B-fragments live in VGPRs, c-state and biases in
// registers, h-state ping-pongs in LDS on t parity -> 1 barrier per layer.
// =============================================================================
__global__ __launch_bounds__(320) void lstm_kernel(
    const _Float16* __restrict__ f1, const _Float16* __restrict__ f2,
    const float* __restrict__ wih0,       // (160,15)
    const float* __restrict__ wih_rest,   // (3,160,40)
    const float* __restrict__ whh,        // (4,160,40)
    const float* __restrict__ bih,        // (4,160)
    const float* __restrict__ bhh,        // (4,160)
    float* __restrict__ hl, float* __restrict__ hr)
{
    extern __shared__ __align__(16) char smem_raw[];
    _Float16* sh = (_Float16*)smem_raw;
    float*    s_scr = (float*)(smem_raw + HALFS_TOTAL * 2);  // [10][16][16]

    _Float16* s_wih = sh + H_WIH;   // [l][n_phys][64]  B-fragment layout, K padded
    _Float16* s_whh = sh + H_WHH;   // [l][n_phys][64]
    _Float16* s_h   = sh + H_H;     // [buf][l][16][64] f16 hidden state, K padded
    _Float16* s_x   = sh + H_X;     // [16][32]         layer-0 input tile

    const int tid  = threadIdx.x;
    const int wave = tid >> 5;      // 0..9 -> gate N-tile (hidden units 4w..4w+3)
    const int lane = tid & 31;
    const int col  = lane & 15;     // nlocal = 4*jl + g
    const int m0   = (lane >> 4) << 3;

    // ---- stage weights (f32 -> f16, gate-interleaved rows, K zero-padded) ----
    for (int i = tid; i < N_WIH; i += 320) { s_wih[i] = (_Float16)0.0f; s_whh[i] = (_Float16)0.0f; }
    for (int i = tid; i < N_H;   i += 320) s_h[i] = (_Float16)0.0f;
    for (int i = tid; i < N_X;   i += 320) s_x[i] = (_Float16)0.0f;
    __syncthreads();
    for (int i = tid; i < 160 * 15; i += 320) {
        int n = i / 15, k = i % 15;
        s_wih[permute_row(n) * 64 + k] = (_Float16)wih0[i];
    }
    for (int i = tid; i < 3 * 160 * 40; i += 320) {
        int l = i / (160 * 40), rr = i % (160 * 40);
        int n = rr / 40, k = rr % 40;
        s_wih[((l + 1) * 160 + permute_row(n)) * 64 + k] = (_Float16)wih_rest[i];
    }
    for (int i = tid; i < 4 * 160 * 40; i += 320) {
        int l = i / (160 * 40), rr = i % (160 * 40);
        int n = rr / 40, k = rr % 40;
        s_whh[(l * 160 + permute_row(n)) * 64 + k] = (_Float16)whh[i];
    }
    __syncthreads();

    // ---- hoist loop-invariant weight B-fragments into VGPRs ----
    v16h wihB[NLAY][2], whhB[NLAY][2];
    #pragma unroll
    for (int l = 0; l < NLAY; ++l) {
        const _Float16* wb = s_wih + (l * 160 + wave * 16) * 64;
        const _Float16* hb = s_whh + (l * 160 + wave * 16) * 64;
        wihB[l][0] = frag16(wb, 64, lane, 0);
        wihB[l][1] = frag16(wb, 64, lane, 32);
        whhB[l][0] = frag16(hb, 64, lane, 0);
        whhB[l][1] = frag16(hb, 64, lane, 32);
    }

    // ---- per-lane fused bias (acc init value; N column fixed per lane) ----
    float bl[NLAY];
    {
        int j = 4 * wave + (col >> 2), g = col & 3;
        int n_orig = g * HID + j;
        #pragma unroll
        for (int l = 0; l < NLAY; ++l)
            bl[l] = bih[l * NG + n_orig] + bhh[l * NG + n_orig];
    }

    // ---- per-lane c-state registers: 2 (m, j) cells per layer ----
    float creg[NLAY][2];
    #pragma unroll
    for (int l = 0; l < NLAY; ++l) { creg[l][0] = 0.0f; creg[l][1] = 0.0f; }

    const int bt = blockIdx.x;                       // batch tile
    const _Float16* xsrc = blockIdx.y ? f2 : f1;
    float* hout          = blockIdx.y ? hr : hl;

    const int xrow = tid >> 4, xk = tid & 15;        // valid when tid < 256
    _Float16 xreg = (_Float16)0.0f;
    if (tid < 256) xreg = xsrc[((size_t)0 * BATCH + bt * 16 + xrow) * 16 + xk];

    float* scr = s_scr + wave * 256;                 // per-wave 16x16 f32 scratch

    for (int t = 0; t < SEQ; ++t) {
        const int cur = t & 1, prv = cur ^ 1;
        if (tid < 256) s_x[xrow * 32 + xk] = xreg;
        __syncthreads();
        // prefetch next timestep's x-tile (latency hidden behind this step)
        if (t + 1 < SEQ && tid < 256)
            xreg = xsrc[((size_t)(t + 1) * BATCH + bt * 16 + xrow) * 16 + xk];

        #pragma unroll
        for (int l = 0; l < NLAY; ++l) {
            v8f acc;
            #pragma unroll
            for (int v = 0; v < 8; ++v) acc[v] = bl[l];
            if (l == 0) {
                acc = wmma_acc(frag16(s_x, 32, lane, 0), wihB[0][0], acc);
            } else {
                const _Float16* xin = s_h + ((cur * 4 + (l - 1)) * 16) * 64;  // h_{l-1,t}
                acc = wmma_acc(frag16(xin, 64, lane, 0),  wihB[l][0], acc);
                acc = wmma_acc(frag16(xin, 64, lane, 32), wihB[l][1], acc);
            }
            const _Float16* hprev = s_h + ((prv * 4 + l) * 16) * 64;          // h_{l,t-1}
            acc = wmma_acc(frag16(hprev, 64, lane, 0),  whhB[l][0], acc);
            acc = wmma_acc(frag16(hprev, 64, lane, 32), whhB[l][1], acc);

            // intra-wave gate exchange through per-wave scratch (no barrier:
            // DS ops from one wave are in-order)
            #pragma unroll
            for (int v = 0; v < 8; ++v) scr[(m0 + v) * 16 + col] = acc[v];
            #pragma unroll
            for (int half = 0; half < 2; ++half) {
                int e = lane + 32 * half;
                int m = e >> 2, jl = e & 3;
                v4f q = *(const v4f*)(scr + m * 16 + 4 * jl);  // i,f,g,o quad
                float nc = fast_sig(q[1]) * creg[l][half] + fast_sig(q[0]) * fast_tanh(q[2]);
                creg[l][half] = nc;
                s_h[((cur * 4 + l) * 16 + m) * 64 + (wave * 4 + jl)] =
                    (_Float16)(fast_sig(q[3]) * fast_tanh(nc));
            }
            __syncthreads();   // publish h_{l,t} for layer l+1 / next timestep
        }
    }

    // final-layer hidden state -> global (last write was into buf (SEQ-1)&1)
    const int fb = (SEQ - 1) & 1;
    for (int e = tid; e < 16 * HID; e += 320) {
        int m = e / HID, j = e % HID;
        hout[(size_t)(bt * 16 + m) * HID + j] = (float)s_h[((fb * 4 + 3) * 16 + m) * 64 + j];
    }
}

// =============================================================================
// Kernel 3: MLP2.  One block per batch row b; hl/hr (80 floats) shared in LDS,
// 256 threads each handle one timestep t of T.
// =============================================================================
__global__ __launch_bounds__(256) void mlp2_kernel(
    const float* __restrict__ T,
    const float* __restrict__ hl, const float* __restrict__ hr,
    const float* __restrict__ w1, const float* __restrict__ b1,
    const float* __restrict__ w2, const float* __restrict__ b2,
    const float* __restrict__ w3, const float* __restrict__ b3,
    const float* __restrict__ w4, const float* __restrict__ b4,
    float* __restrict__ out)
{
    __shared__ float s[5753];
    __shared__ float shl[40], shr[40];
    const int tid = threadIdx.x;
    const int b   = blockIdx.x;

    for (int i = tid; i < 3645; i += 256) s[i]        = w1[i];
    for (int i = tid; i < 45;   i += 256) s[3645 + i] = b1[i];
    for (int i = tid; i < 1350; i += 256) s[3690 + i] = w2[i];
    for (int i = tid; i < 30;   i += 256) s[5040 + i] = b2[i];
    for (int i = tid; i < 600;  i += 256) s[5070 + i] = w3[i];
    for (int i = tid; i < 20;   i += 256) s[5670 + i] = b3[i];
    for (int i = tid; i < 60;   i += 256) s[5690 + i] = w4[i];
    for (int i = tid; i < 3;    i += 256) s[5750 + i] = b4[i];
    if (tid < 40)                 shl[tid]      = hl[(size_t)b * 40 + tid];
    else if (tid < 80)            shr[tid - 40] = hr[(size_t)b * 40 + (tid - 40)];
    __syncthreads();

    const float* sw1 = s;        const float* sb1 = s + 3645;
    const float* sw2 = s + 3690; const float* sb2 = s + 5040;
    const float* sw3 = s + 5070; const float* sb3 = s + 5670;
    const float* sw4 = s + 5690; const float* sb4 = s + 5750;

    const float tv = T[(size_t)b * TLEN + tid];

    float h1[45];
    #pragma unroll 3
    for (int j = 0; j < 45; ++j) {
        const float* wr = sw1 + j * 81;
        float a = sb1[j];
        #pragma unroll
        for (int k = 0; k < 40; ++k) a = fmaf(wr[k], shl[k], a);
        #pragma unroll
        for (int k = 0; k < 40; ++k) a = fmaf(wr[40 + k], shr[k], a);
        a = fmaf(wr[80], tv, a);
        h1[j] = fmaxf(a, 0.0f);
    }
    float h2[30];
    #pragma unroll 3
    for (int j = 0; j < 30; ++j) {
        float a = sb2[j];
        #pragma unroll
        for (int k = 0; k < 45; ++k) a = fmaf(sw2[j * 45 + k], h1[k], a);
        h2[j] = fast_tanh(a);
    }
    float h3[20];
    #pragma unroll
    for (int j = 0; j < 20; ++j) {
        float a = sb3[j];
        #pragma unroll
        for (int k = 0; k < 30; ++k) a = fmaf(sw3[j * 30 + k], h2[k], a);
        h3[j] = fast_tanh(a);
    }
    float* o = out + ((size_t)b * TLEN + tid) * 3;
    #pragma unroll
    for (int j = 0; j < 3; ++j) {
        float a = sb4[j];
        #pragma unroll
        for (int k = 0; k < 20; ++k) a = fmaf(sw4[j * 20 + k], h3[k], a);
        o[j] = a;
    }
}

// =============================================================================
extern "C" void kernel_launch(void* const* d_in, const int* in_sizes, int n_in,
                              void* d_out, int out_size, void* d_ws, size_t ws_size,
                              hipStream_t stream) {
    (void)in_sizes; (void)n_in; (void)out_size; (void)ws_size;
    const float* X1       = (const float*)d_in[0];
    const float* X2       = (const float*)d_in[1];
    const float* T        = (const float*)d_in[2];
    const float* m1w1     = (const float*)d_in[3];
    const float* m1b1     = (const float*)d_in[4];
    const float* m1w2     = (const float*)d_in[5];
    const float* m1b2     = (const float*)d_in[6];
    const float* m1w3     = (const float*)d_in[7];
    const float* m1b3     = (const float*)d_in[8];
    const float* m1w4     = (const float*)d_in[9];
    const float* m1b4     = (const float*)d_in[10];
    const float* wih0     = (const float*)d_in[11];
    const float* wih_rest = (const float*)d_in[12];
    const float* whh      = (const float*)d_in[13];
    const float* bih      = (const float*)d_in[14];
    const float* bhh      = (const float*)d_in[15];
    const float* m2w1     = (const float*)d_in[16];
    const float* m2b1     = (const float*)d_in[17];
    const float* m2w2     = (const float*)d_in[18];
    const float* m2b2     = (const float*)d_in[19];
    const float* m2w3     = (const float*)d_in[20];
    const float* m2b3     = (const float*)d_in[21];
    const float* m2w4     = (const float*)d_in[22];
    const float* m2b4     = (const float*)d_in[23];
    float* out = (float*)d_out;

    // workspace carve: f1/f2 feature seqs (f16, [t][b][16]) + hl/hr (f32)
    _Float16* f1 = (_Float16*)d_ws;
    _Float16* f2 = f1 + (size_t)BATCH * SEQ * 16;
    float*    hl = (float*)(f2 + (size_t)BATCH * SEQ * 16);
    float*    hr = hl + (size_t)BATCH * HID;

    mlp1_kernel<<<dim3(BATCH * SEQ / 256, 2), 256, 0, stream>>>(
        X1, X2, m1w1, m1b1, m1w2, m1b2, m1w3, m1b3, m1w4, m1b4, f1, f2);

    lstm_kernel<<<dim3(BATCH / 16, 2), 320, LSTM_SMEM_BYTES, stream>>>(
        f1, f2, wih0, wih_rest, whh, bih, bhh, hl, hr);

    mlp2_kernel<<<dim3(BATCH), 256, 0, stream>>>(
        T, hl, hr, m2w1, m2b1, m2w2, m2b2, m2w3, m2b3, m2w4, m2b4, out);
}
